// _DeformConvWrapper_24043226923067
// MI455X (gfx1250) — compile-verified, matrix operand order checked by reading, not run on hardware
//
#include <hip/hip_runtime.h>

typedef __attribute__((ext_vector_type(16))) _Float16 v16h;
typedef __attribute__((ext_vector_type(8)))  _Float16 v8h;
typedef __attribute__((ext_vector_type(8)))  float    v8f;

#define BATCH 8
#define CIN   64
#define HH    96
#define WW    96
#define COUT  64
#define KTAP  9
#define CK    (CIN * KTAP)     // 576
#define NPIX  (HH * WW)        // 9216
#define TQ    32               // output pixels per workgroup tile
#define LSTR  584              // padded LDS row stride in halves (16B aligned rows)
#define NTHREADS 288           // 9 wave32: 32 pixels x 9 taps

__global__ __launch_bounds__(NTHREADS, 1)
void dcnv2_wmma_kernel(const float* __restrict__ input,
                       const float* __restrict__ offset,
                       const float* __restrict__ mask,
                       const float* __restrict__ weight,
                       const float* __restrict__ bias,
                       float* __restrict__ out) {
  extern __shared__ _Float16 smem[];
  _Float16* sW = smem;                  // [COUT][LSTR]  f16 weights
  _Float16* sV = smem + COUT * LSTR;    // [TQ][LSTR]    f16 im2col panel (pixel-major)

  const int t    = threadIdx.x;
  const int tile = blockIdx.x;                 // 0 .. BATCH*(NPIX/TQ)-1
  const int b       = tile / (NPIX / TQ);
  const int pixBase = (tile % (NPIX / TQ)) * TQ;

  // ---- Phase 0: fp32 weights -> f16 in LDS (coalesced global reads) ----
  for (int i = t; i < COUT * CK; i += NTHREADS) {
    const int o  = i / CK;
    const int kk = i - o * CK;
    sW[o * LSTR + kk] = (_Float16)weight[i];
  }

  // ---- Phase 1: deformable bilinear gather -> f16 panel ----
  {
    const int q   = t & (TQ - 1);               // pixel within tile
    const int k   = t >> 5;                     // tap 0..8
    const int pix = pixBase + q;
    const int oy  = pix / WW;
    const int ox  = pix - oy * WW;
    const int ky  = k / 3;
    const int kx  = k - ky * 3;

    const float offY = offset[((size_t)b * 2 * KTAP + 2 * k) * NPIX + pix];
    const float offX = offset[((size_t)b * 2 * KTAP + 2 * k + 1) * NPIX + pix];
    const float m    = mask[((size_t)b * KTAP + k) * NPIX + pix];

    const float y   = (float)(oy - 1 + ky) + offY;
    const float x   = (float)(ox - 1 + kx) + offX;
    const float y0f = floorf(y);
    const float x0f = floorf(x);
    const float wy  = y - y0f;
    const float wx  = x - x0f;
    const float y1f = y0f + 1.0f;
    const float x1f = x0f + 1.0f;

    const float vy0 = (y0f >= 0.0f && y0f < (float)HH) ? 1.0f : 0.0f;
    const float vy1 = (y1f >= 0.0f && y1f < (float)HH) ? 1.0f : 0.0f;
    const float vx0 = (x0f >= 0.0f && x0f < (float)WW) ? 1.0f : 0.0f;
    const float vx1 = (x1f >= 0.0f && x1f < (float)WW) ? 1.0f : 0.0f;

    const int iy0 = (int)fminf(fmaxf(y0f, 0.0f), (float)(HH - 1));
    const int iy1 = (int)fminf(fmaxf(y1f, 0.0f), (float)(HH - 1));
    const int ix0 = (int)fminf(fmaxf(x0f, 0.0f), (float)(WW - 1));
    const int ix1 = (int)fminf(fmaxf(x1f, 0.0f), (float)(WW - 1));

    const int i00 = iy0 * WW + ix0;
    const int i01 = iy0 * WW + ix1;
    const int i10 = iy1 * WW + ix0;
    const int i11 = iy1 * WW + ix1;

    const float w00 = (1.0f - wy) * (1.0f - wx) * m * (vy0 * vx0);
    const float w01 = (1.0f - wy) * wx * m * (vy0 * vx1);
    const float w10 = wy * (1.0f - wx) * m * (vy1 * vx0);
    const float w11 = wy * wx * m * (vy1 * vx1);

    const float* __restrict__ ip = input + (size_t)b * CIN * NPIX;
    _Float16* dst = sV + q * LSTR + k;          // panel row kk = c*9 + k, col q
#pragma unroll 4
    for (int c = 0; c < CIN; ++c) {
      const float* __restrict__ p = ip + c * NPIX;
      const float v = w00 * p[i00] + w01 * p[i01] + w10 * p[i10] + w11 * p[i11];
      dst[c * KTAP] = (_Float16)v;
    }
  }

  __syncthreads();

  // ---- Phase 2: WMMA GEMM  out[64 x 32] = W[64 x 576] * V[576 x 32] ----
  const int wave = t >> 5;
  if (wave < 8) {
    const int lane = t & 31;
    const int ln = lane & 15;                   // M (A) / N (B) index
    const int lh = lane >> 4;                   // K-half selector
    const int m0 = (wave >> 1) * 16;            // 4 M tiles
    const int n0 = (wave & 1) * 16;             // 2 N tiles

    const _Float16* aRow = sW + (m0 + ln) * LSTR;
    const _Float16* bRow = sV + (n0 + ln) * LSTR;

    v8f acc = {};
#pragma unroll
    for (int kk = 0; kk < CK; kk += 32) {
      // 16-bit 16x32 fragment layout: halves 0..7 -> K = kk + lh*8 + h,
      //                               halves 8..15 -> K = kk + 16 + lh*8 + h
      const v8h a0 = *(const v8h*)(aRow + kk + lh * 8);
      const v8h a1 = *(const v8h*)(aRow + kk + 16 + lh * 8);
      const v8h b0 = *(const v8h*)(bRow + kk + lh * 8);
      const v8h b1 = *(const v8h*)(bRow + kk + 16 + lh * 8);
      v16h a, bm;
#pragma unroll
      for (int i = 0; i < 8; ++i) {
        a[i]  = a0[i];  a[i + 8]  = a1[i];
        bm[i] = b0[i];  bm[i + 8] = b1[i];
      }
      acc = __builtin_amdgcn_wmma_f32_16x16x32_f16(
          false, a, false, bm, (short)0, acc, false, false);
    }

    // D layout: VGPR r -> M = r + 8*lh, N = ln
    const int pix = pixBase + n0 + ln;
#pragma unroll
    for (int r = 0; r < 8; ++r) {
      const int o = m0 + r + lh * 8;
      out[((size_t)b * COUT + o) * NPIX + pix] = acc[r] + bias[o];
    }
  }
}

extern "C" void kernel_launch(void* const* d_in, const int* in_sizes, int n_in,
                              void* d_out, int out_size, void* d_ws, size_t ws_size,
                              hipStream_t stream) {
  (void)in_sizes; (void)n_in; (void)out_size; (void)d_ws; (void)ws_size;
  const float* input  = (const float*)d_in[0];
  const float* offset = (const float*)d_in[1];
  const float* mask   = (const float*)d_in[2];
  const float* weight = (const float*)d_in[3];
  const float* bias   = (const float*)d_in[4];
  float* out = (float*)d_out;

  const int tiles = BATCH * (NPIX / TQ);                         // 2304 workgroups
  const size_t shmem = (size_t)(COUT + TQ) * LSTR * sizeof(_Float16);  // 112128 B
  dcnv2_wmma_kernel<<<dim3(tiles), dim3(NTHREADS), shmem, stream>>>(
      input, offset, mask, weight, bias, out);
}
